// VQVAE_Trainer_60885456388388
// MI455X (gfx1250) — matrix operand order repaired and checked by reading, not compiled, not run on hardware
//
#include <hip/hip_runtime.h>

// VQ-VAE quantize for MI455X (gfx1250, wave32, WMMA + async global->LDS).
// inputs:  d_in[0] = inputs [16,64,64,64] f32, d_in[1] = cluster_mean [64,1024] f32
// outputs: d_out = [outputs (65536*64 f32) | cluster_index as f32 (65536) | loss (1)]

#define D_DIM      64
#define K_CODES    1024
#define N_SAMPLES  65536
#define CHUNK      128      // codes staged in LDS per iteration
#define LDS_STRIDE 72       // bf16 elems per code row: 144 B = 16B-aligned rows for async B128

// ws layout (float offsets)
#define WS_LOSS  0          // 1 float   : loss accumulator
#define WS_NORM  16         // 1024      : ||m_k||^2
#define WS_CMT   2048       // 65536     : transposed codebook cmT[code][d] (f32, for gather)
#define WS_CBHI  67584      // 32768 f   : packed bf16 hi codebook [code][k], 128 B rows
#define WS_CBLO  100352     // 32768 f   : packed bf16 lo codebook [code][k]

typedef __attribute__((ext_vector_type(16))) __bf16 v16bf;
typedef __attribute__((ext_vector_type(8)))  float  v8f;

union BFrag {
  v16bf v;
  unsigned int   u32[8];
  unsigned short u16[16];
};

__device__ __forceinline__ unsigned short f32_bf16_rne(float f) {
  unsigned int u = __float_as_uint(f);
  unsigned int r = 0x7FFFu + ((u >> 16) & 1u);
  return (unsigned short)((u + r) >> 16);
}
__device__ __forceinline__ float bf16_f32(unsigned short b) {
  return __uint_as_float(((unsigned int)b) << 16);
}

__global__ void vq_prep(const float* __restrict__ cm, float* __restrict__ ws) {
  int t = blockIdx.x * blockDim.x + threadIdx.x;   // 0..65535
  if (t < K_CODES * D_DIM) {
    int c = t >> 6, d = t & 63;                    // t = c*64 + d
    float v = cm[(size_t)d * K_CODES + c];
    ws[WS_CMT + t] = v;                            // cmT[c][d]
    unsigned short h = f32_bf16_rne(v);
    ((unsigned short*)(ws + WS_CBHI))[t] = h;      // cb_hi[c][k=d]
    ((unsigned short*)(ws + WS_CBLO))[t] = f32_bf16_rne(v - bf16_f32(h));
  }
  if (t < K_CODES) {
    float s = 0.f;
    for (int d = 0; d < D_DIM; ++d) { float v = cm[(size_t)d * K_CODES + t]; s += v * v; }
    ws[WS_NORM + t] = s;
  }
  if (t == 0) ws[WS_LOSS] = 0.f;
}

__global__ __launch_bounds__(256) void vq_main(const float* __restrict__ x,
                                               float* __restrict__ ws,
                                               float* __restrict__ out,
                                               float* __restrict__ out_idx) {
  __shared__ __align__(16) unsigned short lds_hi[CHUNK * LDS_STRIDE];
  __shared__ __align__(16) unsigned short lds_lo[CHUNK * LDS_STRIDE];
  __shared__ float lds_norm[CHUNK];
  __shared__ int   lds_widx[8][16];

  const int tid  = threadIdx.x;
  const int wave = tid >> 5;
  const int lane = tid & 31;
  const int half = lane >> 4;      // which 16-lane half
  const int m    = lane & 15;      // row (sample) / col (code) within half
  const int s0   = blockIdx.x * 128 + wave * 16;   // this wave's 16 samples

  // ---- Build A fragments: 16 samples x 32 K per fragment, bf16 hi/lo split ----
  // A 16x32 layout (matches B): lanes 0-15 M=lane, VGPR j -> K pair 2j,2j+1;
  // lanes 16-31 same M, K offset +16.
  BFrag a_hi[2], a_lo[2];
  const float* row = x + (size_t)(s0 + m) * D_DIM;
  for (int kk = 0; kk < 2; ++kk) {
    for (int j = 0; j < 8; ++j) {
      int k = kk * 32 + 16 * half + 2 * j;
      float x0 = row[k], x1 = row[k + 1];
      unsigned short h0 = f32_bf16_rne(x0);
      unsigned short h1 = f32_bf16_rne(x1);
      a_hi[kk].u16[2 * j]     = h0;
      a_hi[kk].u16[2 * j + 1] = h1;
      a_lo[kk].u16[2 * j]     = f32_bf16_rne(x0 - bf16_f32(h0));
      a_lo[kk].u16[2 * j + 1] = f32_bf16_rne(x1 - bf16_f32(h1));
    }
  }

  float bd[8]; int bi[8];
  for (int r = 0; r < 8; ++r) { bd[r] = 3.4e38f; bi[r] = 0; }

  const float* norms = ws + WS_NORM;
  const char*  cb_hi = (const char*)(ws + WS_CBHI);
  const char*  cb_lo = (const char*)(ws + WS_CBLO);

  for (int chunk = 0; chunk < K_CODES / CHUNK; ++chunk) {
    __syncthreads();   // previous chunk fully consumed (WMMA consumed all ds loads)

    // ---- Stage codebook chunk via async global->LDS (ASYNCcnt path) ----
    // 128 code rows x 128 B, packed bf16; LDS rows padded to 144 B (16B aligned).
    {
      const char* srch = cb_hi + (size_t)chunk * CHUNK * 128;   // uniform -> SGPR pair
      const char* srcl = cb_lo + (size_t)chunk * CHUNK * 128;
      for (int i = 0; i < 4; ++i) {
        int e   = i * 256 + tid;
        int n   = e >> 3;                 // code row 0..127
        int seg = e & 7;                  // 16-byte segment within row
        unsigned goff = (unsigned)(n * 128 + seg * 16);
        unsigned dh = (unsigned)(size_t)&lds_hi[n * LDS_STRIDE + seg * 8];
        unsigned dl = (unsigned)(size_t)&lds_lo[n * LDS_STRIDE + seg * 8];
        asm volatile("global_load_async_to_lds_b128 %0, %1, %2"
                     :: "v"(dh), "v"(goff), "s"(srch) : "memory");
        asm volatile("global_load_async_to_lds_b128 %0, %1, %2"
                     :: "v"(dl), "v"(goff), "s"(srcl) : "memory");
      }
    }
    if (tid < CHUNK) lds_norm[tid] = norms[chunk * CHUNK + tid];
    asm volatile("s_wait_asynccnt 0x0" ::: "memory");   // own wave's copies landed
    __syncthreads();                                    // all waves' copies visible

    for (int cg = 0; cg < CHUNK / 16; ++cg) {
      // ---- B fragments: 32 K x 16 codes. VGPR j, lanes 0-15 hold K=2j,2j+1;
      // lanes 16-31 hold K=16+2j,16+2j+1. N = lane&15.
      const int crow = (cg * 16 + m) * LDS_STRIDE;     // even -> dword aligned
      const unsigned int* rh = (const unsigned int*)&lds_hi[crow];
      const unsigned int* rl = (const unsigned int*)&lds_lo[crow];
      BFrag b_hi[2], b_lo[2];
      for (int kk = 0; kk < 2; ++kk) {
        for (int j = 0; j < 8; ++j) {
          int kw = (kk * 32 + 16 * half + 2 * j) >> 1; // dword index in row
          b_hi[kk].u32[j] = rh[kw];
          b_lo[kk].u32[j] = rl[kw];
        }
      }

      // Error-compensated bf16 GEMM: dot ~= hi*hi + hi*lo + lo*hi (f32 accum)
      v8f c = {};
      c = __builtin_amdgcn_wmma_f32_16x16x32_bf16(false, a_hi[0].v, false, b_hi[0].v, (short)0, c, false, false);
      c = __builtin_amdgcn_wmma_f32_16x16x32_bf16(false, a_hi[1].v, false, b_hi[1].v, (short)0, c, false, false);
      c = __builtin_amdgcn_wmma_f32_16x16x32_bf16(false, a_hi[0].v, false, b_lo[0].v, (short)0, c, false, false);
      c = __builtin_amdgcn_wmma_f32_16x16x32_bf16(false, a_hi[1].v, false, b_lo[1].v, (short)0, c, false, false);
      c = __builtin_amdgcn_wmma_f32_16x16x32_bf16(false, a_lo[0].v, false, b_hi[0].v, (short)0, c, false, false);
      c = __builtin_amdgcn_wmma_f32_16x16x32_bf16(false, a_lo[1].v, false, b_hi[1].v, (short)0, c, false, false);

      // C/D layout: lane's code n = lane&15 fixed; VGPR r -> sample m = r + 8*half.
      // argmin score drops ||x||^2 (constant per sample): s = ||m||^2 - 2 x.m
      float mn  = lds_norm[cg * 16 + m];
      int  cidx = chunk * CHUNK + cg * 16 + m;
      for (int r = 0; r < 8; ++r) {
        float s = mn - 2.0f * c[r];
        if (s < bd[r]) { bd[r] = s; bi[r] = cidx; }
      }
    }
  }

  // ---- Cross-lane argmin over the 16 codes within each half (wave32 shuffles) ----
  for (int r = 0; r < 8; ++r) {
    float d = bd[r]; int i = bi[r];
    for (int off = 1; off < 16; off <<= 1) {
      float od = __shfl_xor(d, off, 32);
      int   oi = __shfl_xor(i, off, 32);
      if (od < d || (od == d && oi < i)) { d = od; i = oi; }  // first-index tie-break
    }
    bd[r] = d; bi[r] = i;
  }
  if (m == 0) {
    for (int r = 0; r < 8; ++r) {
      int sm = 8 * half + r;
      lds_widx[wave][sm] = bi[r];
      out_idx[s0 + sm] = (float)bi[r];
    }
  }
  __syncthreads();

  // ---- Gather winning codes (coalesced via transposed codebook in ws),
  //      emit straight-through output, accumulate fused loss ----
  const float* cmT = ws + WS_CMT;
  float acc = 0.f;
  for (int mm = 0; mm < 16; ++mm) {
    int idxm = lds_widx[wave][mm];                   // wave-uniform -> scalar
    const float* qrow = cmT + (size_t)idxm * D_DIM;
    size_t base = (size_t)(s0 + mm) * D_DIM;
    for (int p = 0; p < 2; ++p) {
      int d = lane + p * 32;                         // fully coalesced
      float q  = qrow[d];
      float xv = x[base + d];
      out[base + d] = xv + (q - xv);                 // matches reference STE rounding
      float diff = q - xv;
      acc += diff * diff;
    }
  }
  for (int off = 1; off < 32; off <<= 1) acc += __shfl_xor(acc, off, 32);
  if (lane == 0) atomicAdd(&ws[WS_LOSS], acc);
}

__global__ void vq_finalize(const float* __restrict__ ws, float* __restrict__ loss_out) {
  if (blockIdx.x == 0 && threadIdx.x == 0)
    loss_out[0] = 1.25f * ws[WS_LOSS] / 4194304.0f;  // q_loss + 0.25*e_loss, exact scale
}

extern "C" void kernel_launch(void* const* d_in, const int* in_sizes, int n_in,
                              void* d_out, int out_size, void* d_ws, size_t ws_size,
                              hipStream_t stream) {
  (void)in_sizes; (void)n_in; (void)out_size; (void)ws_size;
  const float* x  = (const float*)d_in[0];
  const float* cm = (const float*)d_in[1];
  float* ws       = (float*)d_ws;
  float* out      = (float*)d_out;
  float* out_idx  = out + (size_t)N_SAMPLES * D_DIM;
  float* loss_out = out_idx + N_SAMPLES;

  vq_prep    <<<256, 256, 0, stream>>>(cm, ws);
  vq_main    <<<N_SAMPLES / 128, 256, 0, stream>>>(x, ws, out, out_idx);
  vq_finalize<<<1, 32, 0, stream>>>(ws, loss_out);
}
